// PSEADPolicyNetwork_33586644254808
// MI455X (gfx1250) — matrix operand order repaired and checked by reading, not compiled, not run on hardware
//
#include <hip/hip_runtime.h>

// Problem constants (from reference)
#define NTOK   16384   // B*S = 8*2048
#define DIM    1024
#define FFDIM  4096
#define DINP   512
#define ACTD   256
#define NLAY   8
#define BATCH  8
#define SEQ    2048

// GEMM tiling
#define BM 128
#define BN 256
#define BK 32

typedef __attribute__((ext_vector_type(16))) __bf16 v16bf;
typedef __attribute__((ext_vector_type(8)))  float  v8f;
typedef __attribute__((ext_vector_type(4)))  int    v4i;

#if defined(__HIP_DEVICE_COMPILE__) && \
    __has_builtin(__builtin_amdgcn_global_load_async_to_lds_b128) && \
    __has_builtin(__builtin_amdgcn_s_wait_asynccnt)
#define ASYNC_LDS 1
#else
#define ASYNC_LDS 0
#endif

__device__ __forceinline__ unsigned short f2bf(float f) {
  unsigned int u = __float_as_uint(f);
  u += 0x7FFFu + ((u >> 16) & 1u);      // round-to-nearest-even
  return (unsigned short)(u >> 16);
}
__device__ __forceinline__ float bf2f(unsigned short h) {
  return __uint_as_float((unsigned int)h << 16);
}

// 16-byte global -> LDS copy: async on gfx1250 (ASYNCcnt), sync fallback.
__device__ __forceinline__ void cp16(const unsigned short* __restrict__ g,
                                     unsigned short* l) {
#if ASYNC_LDS
  __builtin_amdgcn_global_load_async_to_lds_b128(
      (__attribute__((address_space(1))) v4i*)g,
      (__attribute__((address_space(3))) v4i*)l, 0, 0);
#else
  *(uint4*)l = *(const uint4*)g;
#endif
}
__device__ __forceinline__ void cp_wait_all() {
#if ASYNC_LDS
  __builtin_amdgcn_s_wait_asynccnt(0);
#endif
}

// ---------------------------------------------------------------- converts
__global__ __launch_bounds__(256) void cvt_bf16_kernel(
    const float* __restrict__ in, unsigned short* __restrict__ out, int n) {
  int i = blockIdx.x * 256 + threadIdx.x;
  if (i < n) out[i] = f2bf(in[i]);
}

// W [K x N] f32  ->  Wt [N x K] bf16   (blockIdx.z selects matrix in a batch)
__global__ __launch_bounds__(256) void cvt_tr_kernel(
    const float* __restrict__ W, unsigned short* __restrict__ Wt, int K, int N) {
  __shared__ float tile[32][33];
  const size_t mat = (size_t)blockIdx.z * K * N;
  const float* Wp = W + mat;
  unsigned short* Wtp = Wt + mat;
  const int kb = blockIdx.y * 32, nb = blockIdx.x * 32;
  const int tx = threadIdx.x & 31, ty = threadIdx.x >> 5;   // 32 x 8
#pragma unroll
  for (int i = 0; i < 32; i += 8)
    tile[ty + i][tx] = Wp[(size_t)(kb + ty + i) * N + (nb + tx)];
  __syncthreads();
#pragma unroll
  for (int i = 0; i < 32; i += 8)
    Wtp[(size_t)(nb + ty + i) * K + (kb + tx)] = f2bf(tile[tx][ty + i]);
}

// ---------------------------------------------------------------- layernorm
// one 256-thread block per row of DIM=1024; writes bf16
__global__ __launch_bounds__(256) void ln_kernel(
    const float* __restrict__ x, const float* __restrict__ g,
    const float* __restrict__ b, unsigned short* __restrict__ h) {
  __shared__ float red[2][8];
  const int row = blockIdx.x;
  const float* xr = x + (size_t)row * DIM;
  float v[4], s = 0.f, ss = 0.f;
#pragma unroll
  for (int i = 0; i < 4; ++i) {
    v[i] = xr[threadIdx.x + i * 256];
    s += v[i]; ss += v[i] * v[i];
  }
#pragma unroll
  for (int o = 16; o > 0; o >>= 1) { s += __shfl_xor(s, o); ss += __shfl_xor(ss, o); }
  const int wave = threadIdx.x >> 5, lane = threadIdx.x & 31;
  if (lane == 0) { red[0][wave] = s; red[1][wave] = ss; }
  __syncthreads();
  float S = 0.f, SS = 0.f;
#pragma unroll
  for (int w = 0; w < 8; ++w) { S += red[0][w]; SS += red[1][w]; }
  const float mu = S * (1.f / DIM);
  const float var = SS * (1.f / DIM) - mu * mu;
  const float rs = rsqrtf(var + 1e-5f);
  unsigned short* hr = h + (size_t)row * DIM;
#pragma unroll
  for (int i = 0; i < 4; ++i) {
    const int c = threadIdx.x + i * 256;
    hr[c] = f2bf((v[i] - mu) * rs * g[c] + b[c]);
  }
}

// ---------------------------------------------------------------- WMMA GEMM
// C[M,N] = A[M,K](bf16) * Bt[N,K]^T(bf16) (+bias)(+res)(ReLU?) -> f32 / bf16
// 128x256 block tile, BK=32, 8 waves (2 M x 4 N), wave tile 64x64 (4x4 WMMA).
// Double-buffered LDS; tile fill via async global->LDS when available.
template<bool BIAS, bool RES, bool RELU, bool WF32, bool WBF16>
__global__ __launch_bounds__(256) void gemm_bf16_kernel(
    const unsigned short* __restrict__ A,
    const unsigned short* __restrict__ Bt,
    const float* __restrict__ bias,
    const float* __restrict__ res,
    float* __restrict__ Cf,
    unsigned short* __restrict__ Cb,
    int M, int N, int K) {
  // stride 40 bf16 = 80 B rows: conflict-free 16B-aligned fragment reads
  __shared__ unsigned short As[2][BM][40];
  __shared__ unsigned short Bs[2][BN][40];

  const int tid = threadIdx.x;
  const int lane = tid & 31, wave = tid >> 5;
  const int wm = wave & 1, wn = wave >> 1;       // 2 x 4 waves
  const int mBase = blockIdx.y * BM, nBase = blockIdx.x * BN;
  const int halfSel = lane >> 4, lane16 = lane & 15;

  // tile-fill ownership: A: 2x16B per thread, B: 4x16B per thread
  const int aRow = tid >> 1, aCol = (tid & 1) * 16;
  const unsigned short* Ag = A  + (size_t)(mBase + aRow) * K + aCol;
  const unsigned short* Bg = Bt + (size_t)(nBase + tid) * K;

  v8f acc[4][4];
#pragma unroll
  for (int m = 0; m < 4; ++m)
#pragma unroll
    for (int n = 0; n < 4; ++n)
#pragma unroll
      for (int r = 0; r < 8; ++r) acc[m][n][r] = 0.0f;

  auto stage = [&](int buf, int k0) {
    cp16(Ag + k0,     &As[buf][aRow][aCol]);
    cp16(Ag + k0 + 8, &As[buf][aRow][aCol + 8]);
    cp16(Bg + k0,      &Bs[buf][tid][0]);
    cp16(Bg + k0 + 8,  &Bs[buf][tid][8]);
    cp16(Bg + k0 + 16, &Bs[buf][tid][16]);
    cp16(Bg + k0 + 24, &Bs[buf][tid][24]);
  };

  stage(0, 0);
  int buf = 0;
  for (int k0 = 0; k0 < K; k0 += BK, buf ^= 1) {
    cp_wait_all();          // previous fill batch has landed in LDS
    __syncthreads();        // all waves past their reads of the buffer we refill
    if (k0 + BK < K) stage(buf ^ 1, k0 + BK);

    union { uint4 q[2]; v16bf v; } af[4], bfr[4];
#pragma unroll
    for (int m = 0; m < 4; ++m) {
      const int row = wm * 64 + m * 16 + lane16;
      // A frag: e<8 -> k = half*8+e ; e>=8 -> k = 16+half*8+(e-8)
      af[m].q[0] = *(const uint4*)&As[buf][row][halfSel * 8];
      af[m].q[1] = *(const uint4*)&As[buf][row][16 + halfSel * 8];
    }
#pragma unroll
    for (int n = 0; n < 4; ++n) {
      const int col = wn * 64 + n * 16 + lane16;
      // B frag: e<8 -> k = half*8+e ; e>=8 -> k = 16+half*8+(e-8)
      bfr[n].q[0] = *(const uint4*)&Bs[buf][col][halfSel * 8];
      bfr[n].q[1] = *(const uint4*)&Bs[buf][col][16 + halfSel * 8];
    }
#pragma unroll
    for (int m = 0; m < 4; ++m)
#pragma unroll
      for (int n = 0; n < 4; ++n)
        acc[m][n] = __builtin_amdgcn_wmma_f32_16x16x32_bf16(
            false, af[m].v, false, bfr[n].v, (short)0, acc[m][n], false, false);
  }

  // epilogue: lane holds n = lane%16, VGPR r -> m = r + 8*half
#pragma unroll
  for (int n = 0; n < 4; ++n) {
    const int colg = nBase + wn * 64 + n * 16 + lane16;
    float bv = 0.0f;
    if constexpr (BIAS) bv = bias[colg];
#pragma unroll
    for (int m = 0; m < 4; ++m) {
      const int rowg0 = mBase + wm * 64 + m * 16 + halfSel * 8;
#pragma unroll
      for (int r = 0; r < 8; ++r) {
        const size_t idx = (size_t)(rowg0 + r) * N + colg;
        float v = acc[m][n][r] + bv;
        if constexpr (RELU) v = fmaxf(v, 0.0f);
        if constexpr (RES)  v += res[idx];
        if constexpr (WF32)  Cf[idx] = v;
        if constexpr (WBF16) Cb[idx] = f2bf(v);
      }
    }
  }
}

// ---------------------------------------------------------------- attention
// windowed (W=8) attention with C8 cyclic symmetrization.
// 64 threads per (window, head) group; 4 groups per block.
__global__ __launch_bounds__(256) void attn_kernel(
    const unsigned short* __restrict__ q, const unsigned short* __restrict__ k,
    const unsigned short* __restrict__ v, unsigned short* __restrict__ o) {
  __shared__ float sc[4][8][8];
  __shared__ float pr[4][8][8];
  const int g = threadIdx.x >> 6, t = threadIdx.x & 63;
  const int gid = blockIdx.x * 4 + g;          // (b*nw + w)*16 + h
  const int hh = gid & 15;
  const int rowBase = (gid >> 4) * 8;          // b*2048 + w*8
  const int cbase = hh * 64;
  const int i = t >> 3, j = t & 7;

  const unsigned short* qi = q + (size_t)(rowBase + i) * DIM + cbase;
  const unsigned short* kj = k + (size_t)(rowBase + j) * DIM + cbase;
  float s = 0.0f;
#pragma unroll 8
  for (int d = 0; d < 64; ++d) s += bf2f(qi[d]) * bf2f(kj[d]);
  sc[g][i][j] = s * 0.125f;                    // /sqrt(64)
  __syncthreads();
  float sym = 0.0f;
#pragma unroll
  for (int gg = 0; gg < 8; ++gg) sym += sc[g][(i + gg) & 7][(j + gg) & 7];
  sym *= 0.125f;
  pr[g][i][j] = sym;
  __syncthreads();
  float mx = pr[g][i][0];
#pragma unroll
  for (int jj = 1; jj < 8; ++jj) mx = fmaxf(mx, pr[g][i][jj]);
  const float e = __expf(sym - mx);
  sc[g][i][j] = e;
  __syncthreads();
  float den = 0.0f;
#pragma unroll
  for (int jj = 0; jj < 8; ++jj) den += sc[g][i][jj];
  pr[g][i][j] = e / den;
  __syncthreads();
  const int i2 = t >> 3, dc = (t & 7) * 8;
  float ov[8];
#pragma unroll
  for (int dd = 0; dd < 8; ++dd) ov[dd] = 0.0f;
#pragma unroll
  for (int jj = 0; jj < 8; ++jj) {
    const float p = pr[g][i2][jj];
    const unsigned short* vj = v + (size_t)(rowBase + jj) * DIM + cbase + dc;
#pragma unroll
    for (int dd = 0; dd < 8; ++dd) ov[dd] += p * bf2f(vj[dd]);
  }
  unsigned short* oo = o + (size_t)(rowBase + i2) * DIM + cbase + dc;
#pragma unroll
  for (int dd = 0; dd < 8; ++dd) oo[dd] = f2bf(ov[dd]);
}

// ---------------------------------------------------------------- pool + head
__global__ __launch_bounds__(256) void pool_kernel(
    const float* __restrict__ x, float* __restrict__ xm) {
  const int idx = blockIdx.x * 256 + threadIdx.x;  // 0..8191
  const int b = idx >> 10, d = idx & 1023;
  const float* xp = x + (size_t)b * SEQ * DIM + d;
  float s = 0.0f;
  for (int si = 0; si < SEQ; ++si) s += xp[(size_t)si * DIM];
  xm[idx] = s * (1.0f / SEQ);
}

__global__ __launch_bounds__(256) void head_kernel(
    const float* __restrict__ xm, const float* __restrict__ Wout,
    const float* __restrict__ bout, float* __restrict__ out) {
  const int idx = blockIdx.x * 256 + threadIdx.x;  // 0..2047
  const int b = idx >> 8, a = idx & 255;
  const float* xp = xm + (size_t)b * DIM;
  float s = bout[a];
  for (int d = 0; d < DIM; ++d) s += xp[d] * Wout[(size_t)d * ACTD + a];
  out[idx] = s;
}

// ---------------------------------------------------------------- launch
extern "C" void kernel_launch(void* const* d_in, const int* in_sizes, int n_in,
                              void* d_out, int out_size, void* d_ws, size_t ws_size,
                              hipStream_t stream) {
  (void)in_sizes; (void)n_in; (void)out_size; (void)ws_size;
  const float* obs  = (const float*)d_in[0];
  const float* We   = (const float*)d_in[1];
  const float* be   = (const float*)d_in[2];
  const float* Wq   = (const float*)d_in[3];
  const float* bq   = (const float*)d_in[4];
  const float* Wk   = (const float*)d_in[5];
  const float* bk   = (const float*)d_in[6];
  const float* Wv   = (const float*)d_in[7];
  const float* bv   = (const float*)d_in[8];
  const float* Wo   = (const float*)d_in[9];
  const float* bo   = (const float*)d_in[10];
  const float* ln1g = (const float*)d_in[11];
  const float* ln1b = (const float*)d_in[12];
  const float* ln2g = (const float*)d_in[13];
  const float* ln2b = (const float*)d_in[14];
  const float* W1   = (const float*)d_in[15];
  const float* b1   = (const float*)d_in[16];
  const float* W2   = (const float*)d_in[17];
  const float* b2   = (const float*)d_in[18];
  const float* Wout = (const float*)d_in[19];
  const float* bout = (const float*)d_in[20];
  float* out = (float*)d_out;

  char* p = (char*)d_ws;
  auto alloc = [&](size_t bytes) -> char* {
    char* r = p; p += (bytes + 255) & ~(size_t)255; return r;
  };
  unsigned short* obsb  = (unsigned short*)alloc((size_t)NTOK * DINP * 2);
  unsigned short* WeT   = (unsigned short*)alloc((size_t)DINP * DIM * 2);
  unsigned short* WqT   = (unsigned short*)alloc((size_t)NLAY * DIM * DIM * 2);
  unsigned short* WkT   = (unsigned short*)alloc((size_t)NLAY * DIM * DIM * 2);
  unsigned short* WvT   = (unsigned short*)alloc((size_t)NLAY * DIM * DIM * 2);
  unsigned short* WoT   = (unsigned short*)alloc((size_t)NLAY * DIM * DIM * 2);
  unsigned short* W1T   = (unsigned short*)alloc((size_t)NLAY * DIM * FFDIM * 2);
  unsigned short* W2T   = (unsigned short*)alloc((size_t)NLAY * FFDIM * DIM * 2);
  float*          x     = (float*)alloc((size_t)NTOK * DIM * 4);
  unsigned short* h     = (unsigned short*)alloc((size_t)NTOK * DIM * 2);
  unsigned short* qb    = (unsigned short*)alloc((size_t)NTOK * DIM * 2);
  unsigned short* kb    = (unsigned short*)alloc((size_t)NTOK * DIM * 2);
  unsigned short* vb    = (unsigned short*)alloc((size_t)NTOK * DIM * 2);
  unsigned short* attno = (unsigned short*)alloc((size_t)NTOK * DIM * 2);
  unsigned short* hid   = (unsigned short*)alloc((size_t)NTOK * FFDIM * 2);
  float*          xm    = (float*)alloc((size_t)BATCH * DIM * 4);

  // weight / input conversion (bf16, weights transposed to [N x K])
  {
    const int n = NTOK * DINP;
    cvt_bf16_kernel<<<(n + 255) / 256, 256, 0, stream>>>(obs, obsb, n);
  }
  cvt_tr_kernel<<<dim3(DIM / 32, DINP / 32, 1),     256, 0, stream>>>(We, WeT, DINP, DIM);
  cvt_tr_kernel<<<dim3(DIM / 32, DIM / 32, NLAY),   256, 0, stream>>>(Wq, WqT, DIM, DIM);
  cvt_tr_kernel<<<dim3(DIM / 32, DIM / 32, NLAY),   256, 0, stream>>>(Wk, WkT, DIM, DIM);
  cvt_tr_kernel<<<dim3(DIM / 32, DIM / 32, NLAY),   256, 0, stream>>>(Wv, WvT, DIM, DIM);
  cvt_tr_kernel<<<dim3(DIM / 32, DIM / 32, NLAY),   256, 0, stream>>>(Wo, WoT, DIM, DIM);
  cvt_tr_kernel<<<dim3(FFDIM / 32, DIM / 32, NLAY), 256, 0, stream>>>(W1, W1T, DIM, FFDIM);
  cvt_tr_kernel<<<dim3(DIM / 32, FFDIM / 32, NLAY), 256, 0, stream>>>(W2, W2T, FFDIM, DIM);

  const dim3 gD (DIM / BN,   NTOK / BM);   // N=1024  -> (4,128)
  const dim3 gFF(FFDIM / BN, NTOK / BM);   // N=4096  -> (16,128)

  // embed: x = obs @ We + be
  gemm_bf16_kernel<true, false, false, true, false><<<gD, 256, 0, stream>>>(
      obsb, WeT, be, nullptr, x, nullptr, NTOK, DIM, DINP);

  for (int l = 0; l < NLAY; ++l) {
    const size_t wOff  = (size_t)l * DIM * DIM;
    const size_t w1Off = (size_t)l * DIM * FFDIM;

    ln_kernel<<<NTOK, 256, 0, stream>>>(x, ln1g + l * DIM, ln1b + l * DIM, h);

    gemm_bf16_kernel<true, false, false, false, true><<<gD, 256, 0, stream>>>(
        h, WqT + wOff, bq + l * DIM, nullptr, nullptr, qb, NTOK, DIM, DIM);
    gemm_bf16_kernel<true, false, false, false, true><<<gD, 256, 0, stream>>>(
        h, WkT + wOff, bk + l * DIM, nullptr, nullptr, kb, NTOK, DIM, DIM);
    gemm_bf16_kernel<true, false, false, false, true><<<gD, 256, 0, stream>>>(
        h, WvT + wOff, bv + l * DIM, nullptr, nullptr, vb, NTOK, DIM, DIM);

    attn_kernel<<<(NTOK / 8) * 16 / 4, 256, 0, stream>>>(qb, kb, vb, attno);

    // x = x + attno @ Wo + bo
    gemm_bf16_kernel<true, true, false, true, false><<<gD, 256, 0, stream>>>(
        attno, WoT + wOff, bo + l * DIM, x, x, nullptr, NTOK, DIM, DIM);

    ln_kernel<<<NTOK, 256, 0, stream>>>(x, ln2g + l * DIM, ln2b + l * DIM, h);

    // hid = relu(h @ W1 + b1)
    gemm_bf16_kernel<true, false, true, false, true><<<gFF, 256, 0, stream>>>(
        h, W1T + w1Off, b1 + l * FFDIM, nullptr, nullptr, hid, NTOK, FFDIM, DIM);
    // x = x + hid @ W2 + b2
    gemm_bf16_kernel<true, true, false, true, false><<<gD, 256, 0, stream>>>(
        hid, W2T + w1Off, b2 + l * DIM, x, x, nullptr, NTOK, DIM, FFDIM);
  }

  pool_kernel<<<(BATCH * DIM) / 256, 256, 0, stream>>>(x, xm);
  head_kernel<<<(BATCH * ACTD) / 256, 256, 0, stream>>>(xm, Wout, bout, out);
}
